// RNN_352187319191
// MI455X (gfx1250) — matrix operand order repaired and checked by reading, not compiled, not run on hardware
//
#include <hip/hip_runtime.h>
#include <math.h>

// Problem dims (fixed by the reference)
#define Bb   128
#define Tt   512
#define INf  1024
#define Hh   1024
#define APAD 8                    // bf16 row padding: stride 1032 -> 4-dword bank rotation
#define AST  (Hh + APAD)          // padded row stride (elements)

typedef __attribute__((ext_vector_type(16))) __bf16 v16bf;
typedef __attribute__((ext_vector_type(8)))  float  v8f;
typedef __attribute__((ext_vector_type(4)))  unsigned int u32x4_t;
typedef __attribute__((ext_vector_type(8)))  int          i32x8_t;
typedef __attribute__((ext_vector_type(4)))  int          i32x4_t;
typedef __attribute__((ext_vector_type(4)))  int          v4i;

union ABFrag { v16bf v; __bf16 h[16]; uint4 q[2]; };

__device__ __forceinline__ v8f wmma_bf16(v16bf a, v16bf b, v8f c) {
  // D = A(16x32 bf16) x B(32x16 bf16) + C(16x16 f32)
  return __builtin_amdgcn_wmma_f32_16x16x32_bf16(false, a, false, b, (short)0, c,
                                                 false, false);
}

#if __has_builtin(__builtin_amdgcn_tensor_load_to_lds) && __has_builtin(__builtin_amdgcn_s_wait_tensorcnt)
#define HAVE_TDM 1
#endif

#if __has_builtin(__builtin_amdgcn_global_load_async_to_lds_b128) && __has_builtin(__builtin_amdgcn_s_wait_asynccnt)
#define HAVE_ASYNC 1
// builtin signature (decoded from clang diagnostics):
//   (addrspace(1) v4i* gsrc, addrspace(3) v4i* ldst, imm offset, imm cpol)
typedef __attribute__((address_space(1))) v4i gv4i;
typedef __attribute__((address_space(3))) v4i lv4i;
#endif

#ifdef __HIP_DEVICE_COMPILE__
#ifdef HAVE_TDM
#pragma message("CDNA5 probe: TDM tensor_load_to_lds path ENABLED")
#else
#pragma message("CDNA5 probe: TDM builtins missing -> cooperative fallback")
#endif
#ifdef HAVE_ASYNC
#pragma message("CDNA5 probe: global_load_async_to_lds_b128 path ENABLED")
#else
#pragma message("CDNA5 probe: async-to-LDS builtins missing -> direct-load fallback")
#endif
#endif

// ---------------------------------------------------------------------------
// Kernel 0: convert Wx (H x IN) and Wh (H x H) f32 -> bf16 into workspace
// ---------------------------------------------------------------------------
__global__ __launch_bounds__(256) void rnn_prep(const float* __restrict__ wx,
                                                const float* __restrict__ wh,
                                                __bf16* __restrict__ out) {
  size_t i = (size_t)blockIdx.x * blockDim.x + threadIdx.x;
  const size_t n1 = (size_t)Hh * INf;
  if (i < n1)            out[i] = (__bf16)wx[i];
  else if (i < 2 * n1)   out[i] = (__bf16)wh[i - n1];
}

// ---------------------------------------------------------------------------
// Kernel 1: xp = x @ Wx^T + bx   written into y (in-place consumed by scan)
// grid = (M/16, H/512), block = 512 (16 waves, 32 N-columns per wave)
// ---------------------------------------------------------------------------
__global__ __launch_bounds__(512) void rnn_proj(const float* __restrict__ x,
                                                const __bf16* __restrict__ wxb,
                                                const float* __restrict__ bx,
                                                float* __restrict__ y) {
  // LDS: staging f32 tile (64KB) overlapped with padded bf16 A tile (33KB)
  __shared__ __align__(16) unsigned char smem[16 * INf * 4]; // LDS offset 0
  float*  xs = (float*)smem;
  __bf16* xb = (__bf16*)smem;

  const int tid  = threadIdx.x;
  const int wave = tid >> 5, lane = tid & 31;
  const int half = lane >> 4, lrow = lane & 15;
  const int mtile = blockIdx.x;                          // 16-row tile of M=B*T
  const int n0    = blockIdx.y * 512 + wave * 32;        // this wave's N base

  const float* gx = x + (size_t)mtile * 16 * INf;        // contiguous 64KB tile

#ifdef HAVE_TDM
  // --- Tensor Data Mover: DMA the 16x1024 f32 tile into LDS (offset 0) ---
  if (wave == 0) {
    unsigned long long ga = (unsigned long long)gx;
    u32x4_t g0 = (u32x4_t)0;
    g0[0] = 1u;                                          // count=1 (valid, user)
    g0[1] = 0u;                                          // lds_addr = 0 (smem base)
    g0[2] = (unsigned)(ga & 0xFFFFFFFFull);
    g0[3] = (unsigned)((ga >> 32) & 0x01FFFFFFull) | (2u << 30); // type=2 "image"
    i32x8_t g1 = (i32x8_t)0;
    g1[0] = (2 << 16);                                   // data_size=4B
    g1[1] = (INf & 0xFFFF) << 16;                        // tensor_dim0 lo16
    g1[2] = (16 << 16);                                  // dim0 hi=0 | tensor_dim1=16
    g1[3] = (INf & 0xFFFF) << 16;                        // dim1 hi=0 | tile_dim0=1024
    g1[4] = 16;                                          // tile_dim1=16, tile_dim2=0
    g1[5] = INf;                                         // tensor_dim0_stride lo32
    i32x4_t gz = (i32x4_t)0;
#if __clang_major__ >= 23
    __builtin_amdgcn_tensor_load_to_lds(g0, g1, gz, gz, (i32x8_t)0, 0);
#else
    __builtin_amdgcn_tensor_load_to_lds(g0, g1, gz, gz, 0);
#endif
    __builtin_amdgcn_s_wait_tensorcnt(0);
  }
  __syncthreads();
#endif

  // --- one-shot f32 -> padded bf16 conversion (regs bridge the alias) ---
  float f[32];
#ifdef HAVE_TDM
  const float* sp = xs + tid * 32;
#else
  const float* sp = gx + tid * 32;                       // fallback: straight from L2
#endif
#pragma unroll
  for (int i = 0; i < 8; ++i)
    *(float4*)(f + 4 * i) = *(const float4*)(sp + 4 * i);
  __syncthreads();                                       // all reads done before overwrite
  {
    const int row = (tid * 32) >> 10, col = (tid * 32) & 1023;
    union { __bf16 h[32]; uint4 q[4]; } pk;
#pragma unroll
    for (int i = 0; i < 32; ++i) pk.h[i] = (__bf16)f[i];
    uint4* dq = (uint4*)(xb + row * AST + col);          // 16B aligned
#pragma unroll
    for (int i = 0; i < 4; ++i) dq[i] = pk.q[i];
  }
  __syncthreads();

  // --- K loop: 32 x v_wmma_f32_16x16x32_bf16 per output tile ---
  const __bf16* wr0 = wxb + (size_t)(n0 + lrow) * INf;        // B col n0+lrow
  const __bf16* wr1 = wxb + (size_t)(n0 + 16 + lrow) * INf;   // B col n0+16+lrow
  v8f acc0 = {}, acc1 = {};
#pragma unroll 4
  for (int kk = 0; kk < INf; kk += 32) {
    ABFrag a, b0, b1;
    // A-frag: lane<16 -> row=lrow, K = kk+{0..7,16..23}; lane>=16 -> +8
    const __bf16* ar = xb + lrow * AST + kk + half * 8;
    a.q[0] = *(const uint4*)ar;
    a.q[1] = *(const uint4*)(ar + 16);
    // B-frag: same K order as A: K = kk+{0..7,16..23} (+8 for lane>=16)
    const __bf16* p0 = wr0 + kk + half * 8;
    const __bf16* p1 = wr1 + kk + half * 8;
    b0.q[0] = *(const uint4*)p0;  b0.q[1] = *(const uint4*)(p0 + 16);
    b1.q[0] = *(const uint4*)p1;  b1.q[1] = *(const uint4*)(p1 + 16);
    __builtin_prefetch(p0 + 128, 0, 1);
    __builtin_prefetch(p1 + 128, 0, 1);
    acc0 = wmma_bf16(a.v, b0.v, acc0);
    acc1 = wmma_bf16(a.v, b1.v, acc1);
  }

  // --- epilogue: +bias, store xp into y ---
  const float bias0 = bx[n0 + lrow];
  const float bias1 = bx[n0 + 16 + lrow];
  const size_t mbase = (size_t)mtile * 16;
#pragma unroll
  for (int r = 0; r < 8; ++r) {
    const size_t m = mbase + r + half * 8;               // C/D layout: m=r+8*half
    y[m * Hh + n0 + lrow]      = acc0[r] + bias0;
    y[m * Hh + n0 + 16 + lrow] = acc1[r] + bias1;
  }
}

// ---------------------------------------------------------------------------
// Kernel 2: sequential scan  h = tanh(xp_t + h @ Wh^T), y[:,t,:] = h
// grid = B/16 (independent recurrences -> one WGP each, no cross-WG sync),
// block = 1024 (32 waves x 32 N-cols).  Dynamic LDS: 33KB h + 2x64KB xp.
// ---------------------------------------------------------------------------
#define HBUF_BYTES (16 * AST * 2)          // 33024, 16B-aligned
#define XPB_FLOATS (16 * Hh)               // one 64KB xp tile
#define SCAN_LDS   (HBUF_BYTES + 2 * XPB_FLOATS * 4)

__global__ __launch_bounds__(1024) void rnn_scan(const float* __restrict__ h0,
                                                 const __bf16* __restrict__ whb,
                                                 float* __restrict__ y,
                                                 float* __restrict__ hT) {
  extern __shared__ __align__(16) unsigned char dynsmem[];
  __bf16* hbuf = (__bf16*)dynsmem;                       // padded h tile
  float*  xpb  = (float*)(dynsmem + HBUF_BYTES);         // 2 xp tiles (async path)

  const int tid  = threadIdx.x;
  const int wave = tid >> 5, lane = tid & 31;
  const int half = lane >> 4, lrow = lane & 15;
  const int bbase = blockIdx.x * 16;                     // batch rows of this block
  const int n0    = wave * 32;                           // this wave's N base

#ifdef HAVE_ASYNC
  // prime xp(t=0) into buffer 0 while h is initialized
#pragma unroll
  for (int c = 0; c < 4; ++c) {
    const int chunk = tid + c * 1024;                    // 4096 x b128 chunks
    const int row = chunk >> 8, col4 = (chunk & 255) * 4;
    const float* g = y + ((size_t)(bbase + row) * Tt + 0) * Hh + col4;
    __builtin_amdgcn_global_load_async_to_lds_b128(
        (gv4i*)g, (lv4i*)(xpb + row * Hh + col4), 0, 0);
  }
#endif

  // init h tile from h0 (f32 -> bf16)
  for (int i = tid; i < 16 * Hh; i += 1024) {
    const int row = i >> 10, col = i & 1023;
    hbuf[row * AST + col] = (__bf16)h0[(size_t)(bbase + row) * Hh + col];
  }
#ifdef HAVE_ASYNC
  __builtin_amdgcn_s_wait_asynccnt(0);
#endif
  __syncthreads();

  const __bf16* wr0 = whb + (size_t)(n0 + lrow) * Hh;
  const __bf16* wr1 = whb + (size_t)(n0 + 16 + lrow) * Hh;

  for (int t = 0; t < Tt; ++t) {
    // C init = xp tile (phase A result, in place in y)
    v8f acc0 = {}, acc1 = {};
#ifdef HAVE_ASYNC
    const float* xp = xpb + (t & 1) * XPB_FLOATS;
#pragma unroll
    for (int r = 0; r < 8; ++r) {
      acc0[r] = xp[(r + half * 8) * Hh + n0 + lrow];
      acc1[r] = xp[(r + half * 8) * Hh + n0 + 16 + lrow];
    }
    // overlap: async-prefetch xp(t+1) into the other buffer during the K loop.
    // Safe: buffer (t+1)&1 was last read at step t-1, fenced by that step's
    // trailing barrier.
    if (t + 1 < Tt) {
      float* dst = xpb + ((t + 1) & 1) * XPB_FLOATS;
#pragma unroll
      for (int c = 0; c < 4; ++c) {
        const int chunk = tid + c * 1024;
        const int row = chunk >> 8, col4 = (chunk & 255) * 4;
        const float* g = y + ((size_t)(bbase + row) * Tt + (t + 1)) * Hh + col4;
        __builtin_amdgcn_global_load_async_to_lds_b128(
            (gv4i*)g, (lv4i*)(dst + row * Hh + col4), 0, 0);
      }
    }
#else
#pragma unroll
    for (int r = 0; r < 8; ++r) {
      const size_t m = (size_t)(bbase + r + half * 8);
      acc0[r] = y[(m * Tt + t) * Hh + n0 + lrow];
      acc1[r] = y[(m * Tt + t) * Hh + n0 + 16 + lrow];
    }
#endif
    // h @ Wh^T : K=1024 in 32-chunks, Wh streamed from L2 (2MB, resident)
#pragma unroll 4
    for (int kk = 0; kk < Hh; kk += 32) {
      ABFrag a, b0, b1;
      const __bf16* ar = hbuf + lrow * AST + kk + half * 8;
      a.q[0] = *(const uint4*)ar;
      a.q[1] = *(const uint4*)(ar + 16);
      const __bf16* p0 = wr0 + kk + half * 8;
      const __bf16* p1 = wr1 + kk + half * 8;
      b0.q[0] = *(const uint4*)p0;  b0.q[1] = *(const uint4*)(p0 + 16);
      b1.q[0] = *(const uint4*)p1;  b1.q[1] = *(const uint4*)(p1 + 16);
      __builtin_prefetch(p0 + 128, 0, 1);
      __builtin_prefetch(p1 + 128, 0, 1);
      acc0 = wmma_bf16(a.v, b0.v, acc0);
      acc1 = wmma_bf16(a.v, b1.v, acc1);
    }
    // tanh, store y (and hT on last step), stage h_new in regs
    float v0[8], v1[8];
#pragma unroll
    for (int r = 0; r < 8; ++r) {
      v0[r] = tanhf(acc0[r]);
      v1[r] = tanhf(acc1[r]);
      const size_t m = (size_t)(bbase + r + half * 8);
      y[(m * Tt + t) * Hh + n0 + lrow]      = v0[r];
      y[(m * Tt + t) * Hh + n0 + 16 + lrow] = v1[r];
      if (t == Tt - 1) {
        hT[m * Hh + n0 + lrow]      = v0[r];
        hT[m * Hh + n0 + 16 + lrow] = v1[r];
      }
    }
    __syncthreads();                                     // all reads of h done
#pragma unroll
    for (int r = 0; r < 8; ++r) {
      const int m = r + half * 8;
      hbuf[m * AST + n0 + lrow]      = (__bf16)v0[r];
      hbuf[m * AST + n0 + 16 + lrow] = (__bf16)v1[r];
    }
#ifdef HAVE_ASYNC
    __builtin_amdgcn_s_wait_asynccnt(0);                 // xp(t+1) landed in LDS
#endif
    __syncthreads();                                     // h_new + xp(t+1) visible
  }
}

// ---------------------------------------------------------------------------
extern "C" void kernel_launch(void* const* d_in, const int* in_sizes, int n_in,
                              void* d_out, int out_size, void* d_ws, size_t ws_size,
                              hipStream_t stream) {
  (void)in_sizes; (void)n_in; (void)out_size; (void)ws_size;
  const float* x  = (const float*)d_in[0];
  const float* h0 = (const float*)d_in[1];
  const float* Wx = (const float*)d_in[2];
  const float* bx = (const float*)d_in[3];
  const float* Wh = (const float*)d_in[4];

  float* y  = (float*)d_out;
  float* hT = y + (size_t)Bb * Tt * Hh;                  // outputs concatenated

  __bf16* wxb = (__bf16*)d_ws;                           // 2MB
  __bf16* whb = wxb + (size_t)Hh * INf;                  // 2MB

  // gfx1250 WGPs have 320KB LDS; opt in to >64KB dynamic shared for the scan.
  (void)hipFuncSetAttribute((const void*)rnn_scan,
                            hipFuncAttributeMaxDynamicSharedMemorySize, SCAN_LDS);

  rnn_prep<<<dim3((2u * Hh * INf) / 256), 256, 0, stream>>>(Wx, Wh, wxb);
  rnn_proj<<<dim3((Bb * Tt) / 16, Hh / 512), 512, 0, stream>>>(x, wxb, bx, y);
  rnn_scan<<<dim3(Bb / 16), 1024, SCAN_LDS, stream>>>(h0, whb, y, hT);
}
